// RPN_34943853920965
// MI455X (gfx1250) — matrix operand order repaired and checked by reading, not compiled, not run on hardware
//
#include <hip/hip_runtime.h>

// ---------------------------------------------------------------------------
// RPN head for MI455X / gfx1250 (wave32, WMMA, async->LDS staged implicit GEMM).
//   conv3x3(1024->512)+ReLU : 2x8 spatial tiles, halo-padded input, A staged
//     in LDS (40KB) via global_load_async_to_lds_b128, 8 waves x 16x64 slab,
//     v_wmma_f32_16x16x32_f16.
//   heads (18 score + 36 loc -> N=64 GEMM), softmax/decode, bitonic argsort,
//   greedy NMS (LDS bitmask), finalize.
// ---------------------------------------------------------------------------

typedef __attribute__((ext_vector_type(16))) _Float16 v16h;
typedef __attribute__((ext_vector_type(8)))  _Float16 v8h;
typedef __attribute__((ext_vector_type(8)))  float    v8f;

#define H_      38
#define W_      50
#define HW_     1900
#define CIN     1024
#define CMID    512
#define NHEAD   64
#define NANCH   17100
#define NSORT   32768
#define MTILES  119         // ceil(1900/16) for head GEMM
#define PH      40          // padded rows  (-1 .. 38)
#define PW      64          // padded cols  (-1 .. 62), zero-filled
#define YT      19          // 38/2 y-tiles
#define XT      7           // ceil(50/8) x-tiles
#define NMSW    535
#define NMS_THR 0.5f
#define MIN_SZ  16.0f
#define NEG_INF (-1e9f)

// ------------------- async global->LDS copy (guarded) ----------------------
#if defined(__has_builtin)
#if __has_builtin(__builtin_amdgcn_global_load_async_to_lds_b128) && \
    __has_builtin(__builtin_amdgcn_s_wait_asynccnt)
#define HAS_ASYNC_LDS 1
#endif
#endif

#if defined(HAS_ASYNC_LDS)
// Builtin signature (from clang diagnostic): 128-bit payload as v4i pointers,
// global (AS1) source, LDS (AS3) destination, imm offset, imm cpol.
typedef int v4i_ __attribute__((vector_size(16)));
__device__ __forceinline__ void async_cp16(const _Float16* g, _Float16* l) {
  __builtin_amdgcn_global_load_async_to_lds_b128(
      (__attribute__((address_space(1))) v4i_*)(_Float16*)g,
      (__attribute__((address_space(3))) v4i_*)l, 0, 0);
}
__device__ __forceinline__ void async_wait() {
  __builtin_amdgcn_s_wait_asynccnt(0);
}
#else
__device__ __forceinline__ void async_cp16(const _Float16* g, _Float16* l) {
  *(v8h*)l = *(const v8h*)g;
}
__device__ __forceinline__ void async_wait() {}
#endif

// ---------------------------- repack kernels -------------------------------

// feat [1024][38][50] fp32 -> featP [40][64][1024] half, zero halo/pad
__global__ void rpn_pack_feat(const float* __restrict__ feat,
                              _Float16* __restrict__ featP) {
  int i = blockIdx.x * blockDim.x + threadIdx.x;
  if (i >= PH * PW * CIN) return;
  int c = i % CIN;
  int p = i / CIN;
  int px = p % PW, py = p / PW;
  int x = px - 1, y = py - 1;
  float v = 0.f;
  if ((unsigned)x < (unsigned)W_ && (unsigned)y < (unsigned)H_)
    v = feat[(size_t)c * HW_ + y * W_ + x];
  featP[i] = (_Float16)v;
}

// w [512][1024][3][3] fp32 -> wT [tap][cout][cin] half
__global__ void rpn_pack_w3(const float* __restrict__ w,
                            _Float16* __restrict__ wT) {
  int i = blockIdx.x * blockDim.x + threadIdx.x;
  if (i >= 9 * CIN * CMID) return;
  int c = i % CIN;
  int r = i / CIN;
  int o = r % CMID;
  int t = r / CMID;
  wT[i] = (_Float16)w[((size_t)o * CIN + c) * 9 + t];
}

// score_w [18][512], loc_w [36][512] -> headW [n 64][cin 512] half
__global__ void rpn_pack_headw(const float* __restrict__ score_w,
                               const float* __restrict__ loc_w,
                               _Float16* __restrict__ headW) {
  int i = blockIdx.x * blockDim.x + threadIdx.x;
  if (i >= CMID * NHEAD) return;
  int n = i / CMID, c = i % CMID;
  float v = 0.f;
  if (n < 18)      v = score_w[(size_t)n * CMID + c];
  else if (n < 54) v = loc_w[(size_t)(n - 18) * CMID + c];
  headW[i] = (_Float16)v;
}

// ---------------------------- conv3x3 WMMA ---------------------------------
// A 16x32 f16 (wave32): lane l -> M=l&15, g=l>>4; halves 0..7 -> K=g*8+{0..7},
//                       halves 8..15 -> K=16+g*8+{0..7}.
// B 32x16 f16: lane l -> N=l&15, halves same K as A.   C/D f32: VGPR r, lane -> M=r+8*(l>>4),
//                       N=l&15.
// Tile-local M mapping: M -> pixel (y0 + M/8, x0 + M%8).

__global__ void __launch_bounds__(256)
rpn_conv3_wmma(const _Float16* __restrict__ featP,
               const _Float16* __restrict__ wT,
               const float* __restrict__ conv_b,
               _Float16* __restrict__ mid) {
  __shared__ _Float16 smem[40 * 512];           // 4 rows x 10 cols x 512 ch
  int tid  = threadIdx.x;
  int wv   = tid >> 5, lane = tid & 31;
  int ty   = blockIdx.x / XT, tx = blockIdx.x % XT;
  int y0   = ty * 2, x0 = tx * 8;
  int Mloc = lane & 15, group = lane >> 4;
  int my   = Mloc >> 3, mx = Mloc & 7;
  int ob   = wv * 64;

  v8f accs[4] = {{}, {}, {}, {}};

  for (int cc = 0; cc < 2; ++cc) {
    __syncthreads();
    // stage 4x10 halo pixels, 512 channels each (40KB) into LDS
    for (int idx = tid; idx < 2560; idx += 256) {
      int pix = idx >> 6, ch16 = idx & 63;      // 64 x 16B per pixel
      int ly = pix / 10, lx = pix % 10;
      const _Float16* gsrc =
          featP + (size_t)((y0 + ly) * PW + (x0 + lx)) * CIN + cc * 512 + ch16 * 8;
      async_cp16(gsrc, &smem[pix * 512 + ch16 * 8]);
    }
    async_wait();
    __syncthreads();

    for (int t = 0; t < 9; ++t) {
      int dy = t / 3, dx = t % 3;               // halo-local offsets 0..2
      const _Float16* arow = &smem[(size_t)((my + dy) * 10 + (mx + dx)) * 512];
      const _Float16* brow = wT + ((size_t)t * CMID + ob + (lane & 15)) * CIN + cc * 512;
      if (t < 8) __builtin_prefetch(brow + (size_t)CIN * CMID, 0, 3);
#pragma unroll 4
      for (int kk = 0; kk < 16; ++kk) {
        int lc0 = kk * 32;
        union { v16h v; v8h h[2]; } A;
        A.h[0] = *(const v8h*)(arow + lc0 + group * 8);
        A.h[1] = *(const v8h*)(arow + lc0 + 16 + group * 8);
        const _Float16* bk = brow + lc0 + group * 8;
        union { v16h v; v8h h[2]; } B0, B1, B2, B3;
        B0.h[0] = *(const v8h*)(bk);
        B0.h[1] = *(const v8h*)(bk + 16);
        B1.h[0] = *(const v8h*)(bk + (size_t)16 * CIN);
        B1.h[1] = *(const v8h*)(bk + (size_t)16 * CIN + 16);
        B2.h[0] = *(const v8h*)(bk + (size_t)32 * CIN);
        B2.h[1] = *(const v8h*)(bk + (size_t)32 * CIN + 16);
        B3.h[0] = *(const v8h*)(bk + (size_t)48 * CIN);
        B3.h[1] = *(const v8h*)(bk + (size_t)48 * CIN + 16);
        accs[0] = __builtin_amdgcn_wmma_f32_16x16x32_f16(
            false, A.v, false, B0.v, (short)0, accs[0], false, false);
        accs[1] = __builtin_amdgcn_wmma_f32_16x16x32_f16(
            false, A.v, false, B1.v, (short)0, accs[1], false, false);
        accs[2] = __builtin_amdgcn_wmma_f32_16x16x32_f16(
            false, A.v, false, B2.v, (short)0, accs[2], false, false);
        accs[3] = __builtin_amdgcn_wmma_f32_16x16x32_f16(
            false, A.v, false, B3.v, (short)0, accs[3], false, false);
      }
    }
  }

  // store: D VGPR r, lane -> M = r + 8*group -> pixel (y0+group, x0+r)
  int yT = y0 + group;
#pragma unroll
  for (int q = 0; q < 4; ++q) {
    int n = ob + q * 16 + (lane & 15);
    float bias = conv_b[n];
#pragma unroll
    for (int r = 0; r < 8; ++r) {
      int xT = x0 + r;
      if (xT < W_) {
        float v = accs[q][r] + bias;
        v = v > 0.f ? v : 0.f;
        mid[(size_t)(yT * W_ + xT) * CMID + n] = (_Float16)v;
      }
    }
  }
}

// ---------------------------- head WMMA ------------------------------------
// One wave per 16-row M tile computes all N=64 (score+loc fused).

__global__ void rpn_head_wmma(const _Float16* __restrict__ mid,
                              const _Float16* __restrict__ headW,
                              const float* __restrict__ score_b,
                              const float* __restrict__ loc_b,
                              float* __restrict__ heads) {
  int wave = (blockIdx.x * blockDim.x + threadIdx.x) >> 5;
  int lane = threadIdx.x & 31;
  if (wave >= MTILES) return;
  int mbase = wave * 16;
  int row   = mbase + (lane & 15);
  int group = lane >> 4;
  int m = row < HW_ ? row : (HW_ - 1);
  const _Float16* arow = mid + (size_t)m * CMID;

  v8f accs[4] = {{}, {}, {}, {}};
#pragma unroll 4
  for (int kk = 0; kk < 16; ++kk) {
    int c0 = kk * 32;
    union { v16h v; v8h h[2]; } A;
    A.h[0] = *(const v8h*)(arow + c0 + group * 8);
    A.h[1] = *(const v8h*)(arow + c0 + 16 + group * 8);
    const _Float16* bk = headW + (size_t)(lane & 15) * CMID + c0 + group * 8;
    union { v16h v; v8h h[2]; } B0, B1, B2, B3;
    B0.h[0] = *(const v8h*)(bk);
    B0.h[1] = *(const v8h*)(bk + 16);
    B1.h[0] = *(const v8h*)(bk + 16 * CMID);
    B1.h[1] = *(const v8h*)(bk + 16 * CMID + 16);
    B2.h[0] = *(const v8h*)(bk + 32 * CMID);
    B2.h[1] = *(const v8h*)(bk + 32 * CMID + 16);
    B3.h[0] = *(const v8h*)(bk + 48 * CMID);
    B3.h[1] = *(const v8h*)(bk + 48 * CMID + 16);
    accs[0] = __builtin_amdgcn_wmma_f32_16x16x32_f16(
        false, A.v, false, B0.v, (short)0, accs[0], false, false);
    accs[1] = __builtin_amdgcn_wmma_f32_16x16x32_f16(
        false, A.v, false, B1.v, (short)0, accs[1], false, false);
    accs[2] = __builtin_amdgcn_wmma_f32_16x16x32_f16(
        false, A.v, false, B2.v, (short)0, accs[2], false, false);
    accs[3] = __builtin_amdgcn_wmma_f32_16x16x32_f16(
        false, A.v, false, B3.v, (short)0, accs[3], false, false);
  }
#pragma unroll
  for (int q = 0; q < 4; ++q) {
    int n = q * 16 + (lane & 15);
    float bias = (n < 18) ? score_b[n] : ((n < 54) ? loc_b[n - 18] : 0.f);
#pragma unroll
    for (int r = 0; r < 8; ++r) {
      int mr = mbase + r + 8 * group;
      if (mr < HW_) heads[(size_t)mr * NHEAD + n] = accs[q][r] + bias;
    }
  }
}

// ---------------------------- postprocess ----------------------------------

__global__ void rpn_post(const float* __restrict__ heads,
                         const int* __restrict__ stridep,
                         float* __restrict__ out_anch,
                         float* __restrict__ out_loc,
                         float* __restrict__ out_score,
                         float4* __restrict__ rois4,
                         int* __restrict__ validArr,
                         float* __restrict__ keys,
                         int* __restrict__ idxs) {
  int i = blockIdx.x * blockDim.x + threadIdx.x;
  if (i >= NSORT) return;
  if (i >= NANCH) { keys[i] = -3.0e38f; idxs[i] = i; return; }

  int m = i / 9, a = i % 9;
  int x = m % W_, y = m / W_;
  float stride = (float)stridep[0];

  int si = a / 3, ri = a % 3;
  float scale = (si == 0) ? 8.f : (si == 1) ? 16.f : 32.f;
  float ratio = (ri == 0) ? 0.5f : (ri == 1) ? 1.0f : 2.0f;
  float aw = stride * scale * sqrtf(ratio);
  float ah = aw / ratio;
  float ax1 = stride * 0.5f - aw * 0.5f + stride * (float)x;
  float ay1 = stride * 0.5f - ah * 0.5f + stride * (float)y;
  float ax2 = ax1 + aw, ay2 = ay1 + ah;
  out_anch[i * 4 + 0] = ax1; out_anch[i * 4 + 1] = ay1;
  out_anch[i * 4 + 2] = ax2; out_anch[i * 4 + 3] = ay2;

  float s0 = heads[(size_t)m * NHEAD + a * 2];
  float s1 = heads[(size_t)m * NHEAD + a * 2 + 1];
  out_score[i * 2 + 0] = s0;
  out_score[i * 2 + 1] = s1;
  float fg = 1.f / (1.f + __expf(s0 - s1));

  float t0 = heads[(size_t)m * NHEAD + 18 + a * 4 + 0];
  float t1 = heads[(size_t)m * NHEAD + 18 + a * 4 + 1];
  float t2 = heads[(size_t)m * NHEAD + 18 + a * 4 + 2];
  float t3 = heads[(size_t)m * NHEAD + 18 + a * 4 + 3];
  out_loc[i * 4 + 0] = t0; out_loc[i * 4 + 1] = t1;
  out_loc[i * 4 + 2] = t2; out_loc[i * 4 + 3] = t3;

  float ctrx = ax1 + 0.5f * aw, ctry = ay1 + 0.5f * ah;
  float cx = t0 * aw + ctrx, cy = t1 * ah + ctry;
  float bw = aw * __expf(t2), bh = ah * __expf(t3);
  float imw = (float)W_ * stride, imh = (float)H_ * stride;
  float r0 = fminf(fmaxf(cx - 0.5f * bw, 0.f), imw);
  float r1 = fminf(fmaxf(cy - 0.5f * bh, 0.f), imh);
  float r2 = fminf(fmaxf(cx + 0.5f * bw, 0.f), imw);
  float r3 = fminf(fmaxf(cy + 0.5f * bh, 0.f), imh);
  rois4[i] = make_float4(r0, r1, r2, r3);

  int valid = ((r2 - r0) >= MIN_SZ) && ((r3 - r1) >= MIN_SZ);
  validArr[i] = valid;
  keys[i] = valid ? fg : NEG_INF;
  idxs[i] = i;
}

// ---------------------------- bitonic sort ---------------------------------

__global__ void rpn_bitonic(float* __restrict__ key, int* __restrict__ idx,
                            int j, int k) {
  int i = blockIdx.x * blockDim.x + threadIdx.x;
  int ixj = i ^ j;
  if (ixj <= i || ixj >= NSORT) return;
  float ka = key[i], kb = key[ixj];
  int ia = idx[i], ib = idx[ixj];
  bool up = ((i & k) == 0);
  bool b_first = (kb > ka) || (kb == ka && ib < ia);
  bool do_swap = up ? b_first : !b_first;
  if (do_swap) {
    key[i] = kb; key[ixj] = ka;
    idx[i] = ib; idx[ixj] = ia;
  }
}

__global__ void rpn_gather(const int* __restrict__ idxs,
                           const float4* __restrict__ rois4,
                           const int* __restrict__ validArr,
                           float4* __restrict__ sboxes,
                           int* __restrict__ svalid) {
  int i = blockIdx.x * blockDim.x + threadIdx.x;
  if (i >= NANCH) return;
  int s = idxs[i];
  sboxes[i] = rois4[s];
  svalid[i] = validArr[s];
}

// ---------------------------- greedy NMS -----------------------------------

__global__ void rpn_nms(const float4* __restrict__ boxes,
                        const int* __restrict__ svalid,
                        unsigned* __restrict__ keepOut) {
  __shared__ unsigned mask[NMSW];
  int tid = threadIdx.x, nthr = blockDim.x;
  for (int w = tid; w < NMSW; w += nthr) {
    unsigned mw = 0;
    for (int b = 0; b < 32; ++b) {
      int i = w * 32 + b;
      if (i < NANCH && svalid[i]) mw |= (1u << b);
    }
    mask[w] = mw;
  }
  __syncthreads();
  for (int i = 0; i < NANCH - 1; ++i) {
    bool ki = (mask[i >> 5] >> (i & 31)) & 1u;
    if (ki) {
      float4 bi = boxes[i];
      float areai = (bi.z - bi.x) * (bi.w - bi.y);
      for (int j = i + 1 + tid; j < NANCH; j += nthr) {
        if ((mask[j >> 5] >> (j & 31)) & 1u) {
          float4 bj = boxes[j];
          float xx1 = fmaxf(bi.x, bj.x), yy1 = fmaxf(bi.y, bj.y);
          float xx2 = fminf(bi.z, bj.z), yy2 = fminf(bi.w, bj.w);
          float inter = fmaxf(xx2 - xx1, 0.f) * fmaxf(yy2 - yy1, 0.f);
          float areaj = (bj.z - bj.x) * (bj.w - bj.y);
          float iou = inter / (areai + areaj - inter + 1e-9f);
          if (iou > NMS_THR) atomicAnd(&mask[j >> 5], ~(1u << (j & 31)));
        }
      }
    }
    __syncthreads();
  }
  for (int w = tid; w < NMSW; w += nthr) keepOut[w] = mask[w];
}

__global__ void rpn_finalize(const float4* __restrict__ sboxes,
                             const unsigned* __restrict__ keepw,
                             float* __restrict__ out_rois,
                             float* __restrict__ out_keep) {
  int i = blockIdx.x * blockDim.x + threadIdx.x;
  if (i >= NANCH) return;
  bool k = (keepw[i >> 5] >> (i & 31)) & 1u;
  float4 b = sboxes[i];
  out_rois[i * 4 + 0] = k ? b.x : 0.f;
  out_rois[i * 4 + 1] = k ? b.y : 0.f;
  out_rois[i * 4 + 2] = k ? b.z : 0.f;
  out_rois[i * 4 + 3] = k ? b.w : 0.f;
  out_keep[i] = k ? 1.f : 0.f;
}

// ---------------------------- launcher -------------------------------------

extern "C" void kernel_launch(void* const* d_in, const int* in_sizes, int n_in,
                              void* d_out, int out_size, void* d_ws, size_t ws_size,
                              hipStream_t stream) {
  (void)in_sizes; (void)n_in; (void)out_size; (void)ws_size;
  const float* feat    = (const float*)d_in[0];
  const float* conv_w  = (const float*)d_in[1];
  const float* conv_b  = (const float*)d_in[2];
  const float* score_w = (const float*)d_in[3];
  const float* score_b = (const float*)d_in[4];
  const float* loc_w   = (const float*)d_in[5];
  const float* loc_b   = (const float*)d_in[6];
  const int*   stridep = (const int*)d_in[7];

  char* ws = (char*)d_ws;
  size_t off = 0;
  auto carve = [&](size_t bytes) -> void* {
    void* p = ws + off;
    off = (off + bytes + 255) & ~(size_t)255;
    return p;
  };
  _Float16* featP  = (_Float16*)carve((size_t)PH * PW * CIN * 2);  // 5.2 MB
  _Float16* wT     = (_Float16*)carve((size_t)9 * CIN * CMID * 2); // 9.4 MB
  _Float16* mid    = (_Float16*)carve((size_t)HW_ * CMID * 2);     // 1.9 MB
  _Float16* headW  = (_Float16*)carve((size_t)CMID * NHEAD * 2);
  float*    heads  = (float*)carve((size_t)HW_ * NHEAD * 4);
  float4*   rois4  = (float4*)carve((size_t)NANCH * 16);
  int*      validA = (int*)carve((size_t)NANCH * 4);
  float*    keys   = (float*)carve((size_t)NSORT * 4);
  int*      idxs   = (int*)carve((size_t)NSORT * 4);
  float4*   sboxes = (float4*)carve((size_t)NANCH * 16);
  int*      svalid = (int*)carve((size_t)NANCH * 4);
  unsigned* keepw  = (unsigned*)carve((size_t)NMSW * 4);

  float* out       = (float*)d_out;
  float* out_rois  = out;
  float* out_keep  = out + 4 * NANCH;
  float* out_anch  = out + 5 * NANCH;
  float* out_loc   = out + 9 * NANCH;
  float* out_score = out + 13 * NANCH;

  {
    int n = PH * PW * CIN;
    rpn_pack_feat<<<(n + 255) / 256, 256, 0, stream>>>(feat, featP);
  }
  {
    int n = 9 * CIN * CMID;
    rpn_pack_w3<<<(n + 255) / 256, 256, 0, stream>>>(conv_w, wT);
  }
  {
    int n = CMID * NHEAD;
    rpn_pack_headw<<<(n + 255) / 256, 256, 0, stream>>>(score_w, loc_w, headW);
  }

  // conv3x3+ReLU: 19x7 = 133 workgroups, 8 waves each, async->LDS staged A
  rpn_conv3_wmma<<<YT * XT, 256, 0, stream>>>(featP, wT, conv_b, mid);

  // fused heads: 119 waves
  rpn_head_wmma<<<(MTILES * 32 + 255) / 256, 256, 0, stream>>>(
      mid, headW, score_b, loc_b, heads);

  rpn_post<<<NSORT / 256, 256, 0, stream>>>(
      heads, stridep, out_anch, out_loc, out_score, rois4, validA, keys, idxs);

  for (int k = 2; k <= NSORT; k <<= 1)
    for (int j = k >> 1; j > 0; j >>= 1)
      rpn_bitonic<<<NSORT / 256, 256, 0, stream>>>(keys, idxs, j, k);

  rpn_gather<<<(NANCH + 255) / 256, 256, 0, stream>>>(
      idxs, rois4, validA, sboxes, svalid);

  rpn_nms<<<1, 1024, 0, stream>>>(sboxes, svalid, keepw);

  rpn_finalize<<<(NANCH + 255) / 256, 256, 0, stream>>>(
      sboxes, keepw, out_rois, out_keep);
}
